// DyConv1dLinear_51891794870788
// MI455X (gfx1250) — compile-verified, matrix operand order checked by reading, not run on hardware
//
#include <hip/hip_runtime.h>
#include <hip/hip_bf16.h>
#include <math.h>

// ---------------------------------------------------------------------------
// Problem constants (from reference)
// ---------------------------------------------------------------------------
#define BATCH 8
#define TLEN  1024
#define CIN   512
#define HID   2048
#define KW    31
#define HEADS 4
#define TPRIME (TLEN - KW + 1)   // 994 valid output rows
#define TPAD   1056              // y2 rows per batch (>= 1023+30+1, mult of 32)
#define TPADM  1024              // padded M rows (8 tiles of 128) per batch

// GEMM tiling
#define BLK    128               // block tile M and N
#define KTILE  64                // K depth staged per TDM transfer
#define KSTEP  32                // WMMA K depth (bf16)

typedef __attribute__((ext_vector_type(16))) __bf16 v16bf;
typedef __attribute__((ext_vector_type(8)))  __bf16 v8bf;
typedef __attribute__((ext_vector_type(8)))  float  v8f;
typedef __attribute__((ext_vector_type(4)))  unsigned int v4u;
typedef __attribute__((ext_vector_type(8)))  int v8i_;
typedef __attribute__((ext_vector_type(4)))  int v4i_;

// ---------------------------------------------------------------------------
// TDM: 2D tile load global -> LDS.  data_size=2B (bf16).
// D# group0/group1 per CDNA5 ISA 08_async_tensor §8.3/8.4; groups 2/3 zero.
// ---------------------------------------------------------------------------
__device__ __forceinline__ void tdm_load_2d(unsigned lds_off, const void* gaddr,
                                            unsigned tile_d0, unsigned tile_d1,
                                            unsigned tensor_d0, unsigned tensor_d1,
                                            unsigned stride_elems) {
  unsigned long long ga = (unsigned long long)gaddr;
  v4u g0;
  g0[0] = 1u;                                   // count=1, user mode
  g0[1] = lds_off;                              // lds_addr (byte offset)
  g0[2] = (unsigned)ga;                         // global_addr[31:0]
  g0[3] = (unsigned)((ga >> 32) & 0x01FFFFFFu)  // global_addr[56:32]
          | 0x80000000u;                        // type=2 ("image")
  v8i_ g1;
  g1[0] = (int)(1u << 16);                                      // data_size=2B
  g1[1] = (int)((tensor_d0 & 0xFFFFu) << 16);                   // dim0 lo16
  g1[2] = (int)(((tensor_d0 >> 16) & 0xFFFFu) |
                ((tensor_d1 & 0xFFFFu) << 16));                 // dim0 hi / dim1 lo
  g1[3] = (int)(((tensor_d1 >> 16) & 0xFFFFu) |
                ((tile_d0 & 0xFFFFu) << 16));                   // dim1 hi / tile0
  g1[4] = (int)(tile_d1 & 0xFFFFu);                             // tile1, tile2=0
  g1[5] = (int)stride_elems;                                    // dim0_stride lo32
  g1[6] = 0;
  g1[7] = 0;
  v4i_ z4 = {0, 0, 0, 0};
#if defined(__clang_major__) && (__clang_major__ >= 23)
  v8i_ z8 = {0, 0, 0, 0, 0, 0, 0, 0};
  __builtin_amdgcn_tensor_load_to_lds(g0, g1, z4, z4, z8, 0);
#else
  __builtin_amdgcn_tensor_load_to_lds(g0, g1, z4, z4, 0);
#endif
}

__device__ __forceinline__ unsigned lds_offset_of(const void* p) {
  // gfx1250: flat form of an LDS address keeps the LDS byte offset in the
  // low 32 bits (aperture in the high 32) — truncate.
  return (unsigned)(unsigned long long)p;
}

// ---------------------------------------------------------------------------
// LDS fragment load matching the CDNA5 16-bit A/B layout for 16x32 tiles:
// lanes 0-15 hold K=0..7 and 16..23; lanes 16-31 hold K=8..15 and 24..31.
// `ldk` = LDS row stride in elements (tile is row-major, K contiguous).
// ---------------------------------------------------------------------------
__device__ __forceinline__ v16bf load_frag(const __bf16* tile, int lane, int ldk) {
  int row = lane & 15;
  int half = lane >> 4;
  const v8bf* p0 = (const v8bf*)(tile + row * ldk + half * 8);
  const v8bf* p1 = (const v8bf*)(tile + row * ldk + 16 + half * 8);
  v8bf a = *p0, b = *p1;
  v16bf r;
#pragma unroll
  for (int i = 0; i < 8; ++i) { r[i] = a[i]; r[i + 8] = b[i]; }
  return r;
}

__device__ __forceinline__ float selu_f(float v) {
  const float scale = 1.0507009873554805f;
  const float alpha = 1.6732632423543772f;
  return v > 0.f ? scale * v : scale * alpha * (__expf(v) - 1.f);
}

// ---------------------------------------------------------------------------
// Stage 0a: Wc (O,C,K) f32  ->  WcT (K,O,C) bf16
// ---------------------------------------------------------------------------
__global__ void cvt_wc_kernel(const float* __restrict__ Wc, __bf16* __restrict__ WcT) {
  size_t i = (size_t)blockIdx.x * 256 + threadIdx.x;
  if (i >= (size_t)KW * HID * CIN) return;
  int c = (int)(i % CIN);
  size_t r = i / CIN;
  int o = (int)(r % HID);
  int k = (int)(r / HID);
  WcT[i] = (__bf16)Wc[((size_t)o * CIN + c) * KW + k];
}

// Stage 0b: W2 (C,HID) f32 -> bf16, same layout
__global__ void cvt_w2_kernel(const float* __restrict__ W2, __bf16* __restrict__ W2b) {
  size_t i = (size_t)blockIdx.x * 256 + threadIdx.x;
  if (i >= (size_t)CIN * HID) return;
  W2b[i] = (__bf16)W2[i];
}

// ---------------------------------------------------------------------------
// Stage 1: dynamic conv.  One block = one (b,t) row, 128 threads = 4 waves,
// wave h computes the 31 banded-softmax weights for head h, then all threads
// mix the window and emit y2[b][t][c] = dynconv + bias[c] in bf16.
// Rows t >= TLEN are zero padding (so the conv GEMM can over-read).
// ---------------------------------------------------------------------------
__global__ void __launch_bounds__(128)
dynconv_kernel(const float* __restrict__ x, const float* __restrict__ Wl,
               const float* __restrict__ bl, const float* __restrict__ bias,
               __bf16* __restrict__ y2) {
  int b = blockIdx.x / TPAD;
  int t = blockIdx.x % TPAD;
  int tid = threadIdx.x;
  __bf16* yrow = y2 + ((size_t)b * TPAD + t) * CIN;
  if (t >= TLEN) {
#pragma unroll
    for (int i = 0; i < 4; ++i) yrow[tid + 128 * i] = (__bf16)0.0f;
    return;
  }
  __shared__ float xrow[CIN];
  __shared__ float wsm[HEADS * KW];
  const float* xr = x + ((size_t)b * TLEN + t) * CIN;
#pragma unroll
  for (int i = 0; i < 4; ++i) xrow[tid + 128 * i] = xr[tid + 128 * i];
  __syncthreads();

  int h = tid >> 5;
  int lane = tid & 31;
  float logit = -INFINITY;
  int pos = t + lane - (KW - 1) / 2;
  if (lane < KW && pos >= 0 && pos < TLEN) {
    const float* wl = Wl + (size_t)(h * KW + lane) * CIN;
    float acc = bl[h * KW + lane];
    for (int c = 0; c < CIN; ++c) acc = fmaf(xrow[c], wl[c], acc);
    logit = acc;
  }
  float m = logit;
#pragma unroll
  for (int s = 16; s; s >>= 1) m = fmaxf(m, __shfl_xor(m, s, 32));
  float e = (logit == -INFINITY) ? 0.f : __expf(logit - m);
  float sum = e;
#pragma unroll
  for (int s = 16; s; s >>= 1) sum += __shfl_xor(sum, s, 32);
  if (lane < KW) wsm[h * KW + lane] = e / sum;
  __syncthreads();

#pragma unroll
  for (int i = 0; i < 4; ++i) {
    int c = tid + 128 * i;
    int hh = c >> 7;                       // c / (CIN/HEADS)
    float acc2 = 0.f;
    for (int k = 0; k < KW; ++k) {
      int p = t + k - (KW - 1) / 2;
      p = min(max(p, 0), TLEN - 1);
      acc2 = fmaf(wsm[hh * KW + k], x[((size_t)b * TLEN + p) * CIN + c], acc2);
    }
    yrow[c] = (__bf16)(acc2 + bias[c]);
  }
}

// ---------------------------------------------------------------------------
// Shared GEMM compute: one staged 128x64 A tile + 128x64 B tile -> 16 WMMAs
// per wave (2 sub-k-steps of 8).  Wave grid 4(M) x 2(N); wave tile 32x64.
// ---------------------------------------------------------------------------
__device__ __forceinline__ void wmma_tile_128x128x64(const __bf16* tA, const __bf16* tB,
                                                     int wid, int lane, v8f acc[2][4]) {
  int rb = (wid >> 1) * 32;
  int cb = (wid & 1) * 64;
#pragma unroll
  for (int kk = 0; kk < KTILE / KSTEP; ++kk) {
    const __bf16* pA = tA + kk * KSTEP;
    const __bf16* pB = tB + kk * KSTEP;
    v16bf af[2], bfv[4];
#pragma unroll
    for (int mi = 0; mi < 2; ++mi) af[mi] = load_frag(pA + (rb + mi * 16) * KTILE, lane, KTILE);
#pragma unroll
    for (int ni = 0; ni < 4; ++ni) bfv[ni] = load_frag(pB + (cb + ni * 16) * KTILE, lane, KTILE);
#pragma unroll
    for (int mi = 0; mi < 2; ++mi)
#pragma unroll
      for (int ni = 0; ni < 4; ++ni)
        acc[mi][ni] = __builtin_amdgcn_wmma_f32_16x16x32_bf16(
            false, af[mi], false, bfv[ni], (short)0, acc[mi][ni], false, false);
  }
}

// ---------------------------------------------------------------------------
// Stage 2: conv1d(C->HID, K=31, VALID) as 31 shifted WMMA GEMMs + SELU.
// Grid = B * (TPADM/BLK) * (HID/BLK) blocks, 256 threads (8 waves).
// A tile: y2[b][row0+k .. +127][cc..cc+63]  (128x64 bf16, TDM by wave 0)
// B tile: WcT[k][col0 .. +127][cc..cc+63]   (128x64 bf16, TDM by wave 1)
// ---------------------------------------------------------------------------
__global__ void __launch_bounds__(256)
convgemm_kernel(const __bf16* __restrict__ y2, const __bf16* __restrict__ WcT,
                const float* __restrict__ bc, __bf16* __restrict__ sbuf) {
  __shared__ __bf16 sA[2][BLK * KTILE];
  __shared__ __bf16 sB[2][BLK * KTILE];

  int ntile = blockIdx.x % (HID / BLK);
  int mtile = (blockIdx.x / (HID / BLK)) % (TPADM / BLK);
  int b     = blockIdx.x / ((HID / BLK) * (TPADM / BLK));
  int row0 = mtile * BLK, col0 = ntile * BLK;
  int tid = threadIdx.x, wid = tid >> 5, lane = tid & 31;

  v8f acc[2][4] = {};
  const int CCHUNKS = CIN / KTILE;           // 8
  const int NSTEPS  = KW * CCHUNKS;          // 248

  const __bf16* aBase = y2  + ((size_t)b * TPAD + row0) * CIN;
  const __bf16* bBase = WcT + (size_t)col0 * CIN;

  auto stageA = [&](int buf, int step) {
    int k  = step >> 3;                      // step / CCHUNKS
    int cc = (step & (CCHUNKS - 1)) * KTILE;
    tdm_load_2d(lds_offset_of(&sA[buf][0]), aBase + (size_t)k * CIN + cc,
                KTILE, BLK, CIN, TPAD * BATCH, CIN);
  };
  auto stageB = [&](int buf, int step) {
    int k  = step >> 3;
    int cc = (step & (CCHUNKS - 1)) * KTILE;
    tdm_load_2d(lds_offset_of(&sB[buf][0]), bBase + (size_t)k * (HID * CIN) + cc,
                KTILE, BLK, CIN, KW * HID, CIN);
  };

  int buf = 0;
  if (wid == 0) stageA(0, 0);
  if (wid == 1) stageB(0, 0);
  for (int step = 0; step < NSTEPS; ++step) {
    if (wid <= 1) __builtin_amdgcn_s_wait_tensorcnt(0);
    __syncthreads();
    if (step + 1 < NSTEPS) {
      if (wid == 0) stageA(buf ^ 1, step + 1);
      if (wid == 1) stageB(buf ^ 1, step + 1);
    }
    wmma_tile_128x128x64(&sA[buf][0], &sB[buf][0], wid, lane, acc);
    __syncthreads();
    buf ^= 1;
  }

  // epilogue: selu(z + bc) -> bf16 activation buffer (B, TPADM, HID)
  int mrow = row0 + (wid >> 1) * 32;
  int ncol = col0 + (wid & 1) * 64;
#pragma unroll
  for (int mi = 0; mi < 2; ++mi)
#pragma unroll
    for (int ni = 0; ni < 4; ++ni)
#pragma unroll
      for (int e = 0; e < 8; ++e) {
        int r = mrow + mi * 16 + (lane >> 4) * 8 + e;
        int cg = ncol + ni * 16 + (lane & 15);
        if (r < TPRIME) {
          float v = selu_f(acc[mi][ni][e] + bc[cg]);
          sbuf[((size_t)b * TPADM + r) * HID + cg] = (__bf16)v;
        }
      }
}

// ---------------------------------------------------------------------------
// Stage 3: out = selu_act @ W2^T + b2  (M=994/b, N=512, K=2048), f32 output.
// B-tile comes straight from W2 (C,HID) row-major: rows are output channels,
// K (=HID) contiguous — same 128x64 tile shape as A.
// ---------------------------------------------------------------------------
__global__ void __launch_bounds__(256)
outgemm_kernel(const __bf16* __restrict__ sbuf, const __bf16* __restrict__ W2b,
               const float* __restrict__ b2, float* __restrict__ out) {
  __shared__ __bf16 sA[2][BLK * KTILE];
  __shared__ __bf16 sB[2][BLK * KTILE];

  int ntile = blockIdx.x % (CIN / BLK);                 // 4
  int mtile = (blockIdx.x / (CIN / BLK)) % (TPADM / BLK);
  int b     = blockIdx.x / ((CIN / BLK) * (TPADM / BLK));
  int row0 = mtile * BLK, col0 = ntile * BLK;
  int tid = threadIdx.x, wid = tid >> 5, lane = tid & 31;

  v8f acc[2][4] = {};
  const int NSTEPS = HID / KTILE;                       // 32

  const __bf16* aBase = sbuf + ((size_t)b * TPADM + row0) * HID;
  const __bf16* bBase = W2b  + (size_t)col0 * HID;

  auto stageA = [&](int buf, int step) {
    tdm_load_2d(lds_offset_of(&sA[buf][0]), aBase + step * KTILE,
                KTILE, BLK, HID, TPADM * BATCH, HID);
  };
  auto stageB = [&](int buf, int step) {
    tdm_load_2d(lds_offset_of(&sB[buf][0]), bBase + step * KTILE,
                KTILE, BLK, HID, CIN, HID);
  };

  int buf = 0;
  if (wid == 0) stageA(0, 0);
  if (wid == 1) stageB(0, 0);
  for (int step = 0; step < NSTEPS; ++step) {
    if (wid <= 1) __builtin_amdgcn_s_wait_tensorcnt(0);
    __syncthreads();
    if (step + 1 < NSTEPS) {
      if (wid == 0) stageA(buf ^ 1, step + 1);
      if (wid == 1) stageB(buf ^ 1, step + 1);
    }
    wmma_tile_128x128x64(&sA[buf][0], &sB[buf][0], wid, lane, acc);
    __syncthreads();
    buf ^= 1;
  }

  int mrow = row0 + (wid >> 1) * 32;
  int ncol = col0 + (wid & 1) * 64;
#pragma unroll
  for (int mi = 0; mi < 2; ++mi)
#pragma unroll
    for (int ni = 0; ni < 4; ++ni)
#pragma unroll
      for (int e = 0; e < 8; ++e) {
        int r = mrow + mi * 16 + (lane >> 4) * 8 + e;
        int cg = ncol + ni * 16 + (lane & 15);
        if (r < TPRIME) {
          out[((size_t)b * TPRIME + r) * CIN + cg] = acc[mi][ni][e] + b2[cg];
        }
      }
}

// ---------------------------------------------------------------------------
// Launch
// ---------------------------------------------------------------------------
extern "C" void kernel_launch(void* const* d_in, const int* in_sizes, int n_in,
                              void* d_out, int out_size, void* d_ws, size_t ws_size,
                              hipStream_t stream) {
  const float* x    = (const float*)d_in[0];
  const float* Wl   = (const float*)d_in[1];
  const float* bl   = (const float*)d_in[2];
  const float* bias = (const float*)d_in[3];
  const float* Wc   = (const float*)d_in[4];
  const float* bc   = (const float*)d_in[5];
  const float* W2   = (const float*)d_in[6];
  const float* b2   = (const float*)d_in[7];
  float* out = (float*)d_out;

  // workspace partition (bf16 buffers), 256B aligned
  char* ws = (char*)d_ws;
  auto alignup = [](size_t v) { return (v + 255) & ~(size_t)255; };
  __bf16* y2   = (__bf16*)ws;                       size_t o0 = alignup((size_t)BATCH * TPAD * CIN * 2);
  __bf16* WcT  = (__bf16*)(ws + o0);                size_t o1 = o0 + alignup((size_t)KW * HID * CIN * 2);
  __bf16* sbuf = (__bf16*)(ws + o1);                size_t o2 = o1 + alignup((size_t)BATCH * TPADM * HID * 2);
  __bf16* W2b  = (__bf16*)(ws + o2);
  (void)ws_size; (void)n_in; (void)in_sizes; (void)out_size;

  // stage 0: weight conversion / transpose to bf16
  {
    size_t n = (size_t)KW * HID * CIN;
    cvt_wc_kernel<<<(unsigned)((n + 255) / 256), 256, 0, stream>>>(Wc, WcT);
  }
  {
    size_t n = (size_t)CIN * HID;
    cvt_w2_kernel<<<(unsigned)((n + 255) / 256), 256, 0, stream>>>(W2, W2b);
  }
  // stage 1: dynamic conv -> y2 (bf16, zero-padded rows)
  dynconv_kernel<<<BATCH * TPAD, 128, 0, stream>>>(x, Wl, bl, bias, y2);
  // stage 2: conv-as-GEMM + SELU -> sbuf (bf16)
  convgemm_kernel<<<BATCH * (TPADM / BLK) * (HID / BLK), 256, 0, stream>>>(y2, WcT, bc, sbuf);
  // stage 3: output GEMM -> out (f32)
  outgemm_kernel<<<BATCH * (TPADM / BLK) * (CIN / BLK), 256, 0, stream>>>(sbuf, W2b, b2, out);
}
